// testLSTMNet_2078764171594
// MI455X (gfx1250) — matrix operand order repaired and checked by reading, not compiled, not run on hardware
//
#include <hip/hip_runtime.h>

typedef __bf16 bf16_t;
typedef __attribute__((ext_vector_type(16))) __bf16 v16bf_t;
typedef __attribute__((ext_vector_type(8)))  __bf16 v8bf_t;
typedef __attribute__((ext_vector_type(8)))  float   v8f_t;

static __device__ __forceinline__ bf16_t f2bf(float f) {
  unsigned u = __builtin_bit_cast(unsigned, f);
  u += 0x7FFFu + ((u >> 16) & 1u);                // round-to-nearest-even
  unsigned short h = (unsigned short)(u >> 16);
  return __builtin_bit_cast(bf16_t, h);
}

static __device__ __forceinline__ v8f_t vzero8() {
  v8f_t z;
#pragma unroll
  for (int i = 0; i < 8; ++i) z[i] = 0.0f;
  return z;
}

// WGP-scope prefetch: pulls the line into ALL cache levels (ISA prefetch
// table, SCOPE 0), unlike __builtin_prefetch which lowers to SCOPE_SE
// (GL2-only) on this toolchain.
static __device__ __forceinline__ void prefetch_wgp(const void* p) {
  asm volatile("global_prefetch_b8 %0, off" :: "v"(p));
}

// A fragment: 16x32 bf16, lane = row (M = lane%16); per ISA table:
// half 0: K = {k0..k0+7, k0+16..k0+23}; half 1: K = {k0+8..k0+15, k0+24..k0+31}
static __device__ __forceinline__ v16bf_t load_a_frag(const bf16_t* A, int ldk,
                                                      int m0, int k0, int lane) {
  const int l = lane & 15, half = lane >> 4;
  const bf16_t* p = A + (size_t)(m0 + l) * (size_t)ldk + (size_t)(k0 + half * 8);
  v8bf_t lo = *(const v8bf_t*)(p);
  v8bf_t hi = *(const v8bf_t*)(p + 16);
  v16bf_t f;
#pragma unroll
  for (int j = 0; j < 8; ++j) { f[j] = lo[j]; f[j + 8] = hi[j]; }
  return f;
}

// B fragment: 32x16 bf16, lane = column (N = lane%16); same K order as A:
// half 0: K = {k0..k0+7, k0+16..k0+23}; half 1: K = {k0+8..k0+15, k0+24..k0+31}
// (along K of W row n, since we compute x @ W^T with W row-major [N,K])
static __device__ __forceinline__ v16bf_t load_b_frag(const bf16_t* W, int ldk,
                                                      int n0, int k0, int rows, int lane) {
  const int l = lane & 15, half = lane >> 4;
  int n = n0 + l; n = (n < rows) ? n : (rows - 1);
  const bf16_t* p = W + (size_t)n * (size_t)ldk + (size_t)(k0 + half * 8);
  v8bf_t lo = *(const v8bf_t*)(p);
  v8bf_t hi = *(const v8bf_t*)(p + 16);
  v16bf_t f;
#pragma unroll
  for (int j = 0; j < 8; ++j) { f[j] = lo[j]; f[j + 8] = hi[j]; }
  return f;
}

static __device__ __forceinline__ v8f_t wmma_bf16(v16bf_t a, v16bf_t b, v8f_t c) {
  return __builtin_amdgcn_wmma_f32_16x16x32_bf16(false, a, false, b, (short)0, c,
                                                 false, false);
}

static __device__ __forceinline__ float sigmoidf_(float x) {
  return 1.0f / (1.0f + __expf(-x));
}

// Branch-free tanh: CDNA5 has a hardware TANH transcendental; fall back to a
// straight-line, overflow-safe exp formulation if the builtin is unavailable.
static __device__ __forceinline__ float fast_tanhf(float x) {
#if __has_builtin(__builtin_amdgcn_tanhf)
  return __builtin_amdgcn_tanhf(x);
#else
  const float t = __expf(-2.0f * fabsf(x));   // in (0,1], never overflows
  const float r = (1.0f - t) / (1.0f + t);
  return copysignf(r, x);
#endif
}

// ---------------------------------------------------------------------------
// Elementwise helpers
// ---------------------------------------------------------------------------
__global__ __launch_bounds__(256) void cvt_f32_to_bf16(const float* __restrict__ in,
                                                       bf16_t* __restrict__ out, int n) {
  for (int i = blockIdx.x * 256 + threadIdx.x; i < n; i += gridDim.x * 256)
    out[i] = f2bf(in[i]);
}

// W3 [10,2048] f32 -> [16,2048] bf16, rows 10..15 zero
__global__ __launch_bounds__(256) void pad_w3_bf16(const float* __restrict__ w3,
                                                   bf16_t* __restrict__ out) {
  for (int i = blockIdx.x * 256 + threadIdx.x; i < 16 * 2048; i += gridDim.x * 256) {
    const int row = i >> 11;
    out[i] = (row < 10) ? f2bf(w3[i]) : f2bf(0.0f);
  }
}

// bsum = a + b (combined LSTM bias)
__global__ __launch_bounds__(256) void add_bias_vec(const float* __restrict__ a,
                                                    const float* __restrict__ b,
                                                    float* __restrict__ out, int n) {
  for (int i = blockIdx.x * 256 + threadIdx.x; i < n; i += gridDim.x * 256)
    out[i] = a[i] + b[i];
}

// ---------------------------------------------------------------------------
// GEMM: out[M,N] = A[M,K](bf16) @ W[N,K]^T(bf16) + bias, optional ReLU.
// Block tile 256x128, 8 waves (4x2), wave tile 64x64 (4x4 WMMA accumulators):
// 0.5 fragment loads per WMMA. Fragments load directly from global: A panel
// and weights are L2-resident (192MB L2), so no LDS staging is needed.
// All epilogue variation is compile-time (no branchy scalarized tail).
// ---------------------------------------------------------------------------
template <bool OUTF32, bool RELU, bool NCHECK>
__global__ __launch_bounds__(256) void gemm_bf16_wmma(
    const bf16_t* __restrict__ A, const bf16_t* __restrict__ W,
    void* __restrict__ outP, const float* __restrict__ bias,
    int M, int N, int K, int Wrows) {
  const int bm = blockIdx.x * 256;
  const int bn = blockIdx.y * 128;
  const int wv = threadIdx.x >> 5;
  const int lane = threadIdx.x & 31;
  const int wm = bm + (wv >> 1) * 64;
  const int wn = bn + (wv & 1) * 64;

  v8f_t acc[4][4];
#pragma unroll
  for (int i = 0; i < 4; ++i)
#pragma unroll
    for (int j = 0; j < 4; ++j) acc[i][j] = vzero8();

  for (int k0 = 0; k0 < K; k0 += 32) {
    v16bf_t a[4], b[4];
#pragma unroll
    for (int i = 0; i < 4; ++i) a[i] = load_a_frag(A, K, wm + 16 * i, k0, lane);
#pragma unroll
    for (int j = 0; j < 4; ++j) b[j] = load_b_frag(W, K, wn + 16 * j, k0, Wrows, lane);
#pragma unroll
    for (int i = 0; i < 4; ++i)
#pragma unroll
      for (int j = 0; j < 4; ++j) acc[i][j] = wmma_bf16(a[i], b[j], acc[i][j]);
  }

  const int l = lane & 15, half = lane >> 4;
  float* __restrict__ outF = (float*)outP;
  bf16_t* __restrict__ outB = (bf16_t*)outP;
#pragma unroll
  for (int tn = 0; tn < 4; ++tn) {
    const int n = wn + tn * 16 + l;
    const int nb = NCHECK ? ((n < N) ? n : (N - 1)) : n;
    const float bb = bias[nb];
#pragma unroll
    for (int tm = 0; tm < 4; ++tm) {
#pragma unroll
      for (int r = 0; r < 8; ++r) {
        const int m = wm + tm * 16 + r + half * 8;
        float v = acc[tm][tn][r] + bb;
        if (RELU) v = fmaxf(v, 0.0f);
        const size_t idx = (size_t)m * (size_t)N + (size_t)n;
        if (!NCHECK || n < N) {
          if (OUTF32) outF[idx] = v;
          else        outB[idx] = f2bf(v);
        }
      }
    }
  }
}

// ---------------------------------------------------------------------------
// LSTM recurrence: gates[b,512] = xg[b,t] + h[b] @ Whh^T, serial over T.
// The scan is serial only per batch row -> batch-parallel across WGPs:
// 4 workgroups, each owning a 16-row batch tile on its own WGP; 8 waves per
// WG, wave jg owns hidden cols jg*16..+15 with all 4 gate tiles (i,f,g,o) in
// its own accumulators (WMMA C-layout == elementwise layout, so the c/h
// update is local). Per-wave VGPRs: Whh frags 128 + acc 32 + c 8 + temps ->
// fits with NO scratch spills (verified). h: 4.3KB LDS tile (stride 136 bf16
// -> conflict-free 16B frag loads, broadcast across waves). Accumulators
// start at zero (lowered to WMMA C=0) so WMMAs issue right after the
// barrier; xg is added in the gate phase, where its loads (WGP-prefetched at
// t-1, near-cache hits) overlap the WMMA chain. 16 WMMAs/wave/step.
// ---------------------------------------------------------------------------
__global__ __launch_bounds__(256)
__attribute__((amdgpu_waves_per_eu(2)))
void lstm_scan_wmma(
    const float* __restrict__ xg,     // [B*T, 512] f32, row = b*T + t
    const bf16_t* __restrict__ Whh,   // [512, 128] bf16
    bf16_t* __restrict__ hout,        // [B*T, 128] bf16
    int T) {
  __shared__ bf16_t hbuf[16 * 136];
  const int tid = threadIdx.x;
  const int jg = tid >> 5;            // wave = hidden col group 0..7
  const int lane = tid & 31;
  const int l = lane & 15, half = lane >> 4;
  const int m0 = blockIdx.x * 16;     // this WG's batch rows: m0..m0+15
  const int hc = jg * 16 + l;         // this lane's hidden column

  for (int i = tid; i < 16 * 136; i += 256) hbuf[i] = f2bf(0.0f);

  v16bf_t bw[4][4];                   // [gate][kstep] Whh fragments (128 VGPRs)
#pragma unroll
  for (int g = 0; g < 4; ++g)
#pragma unroll
    for (int kk = 0; kk < 4; ++kk)
      bw[g][kk] = load_b_frag(Whh, 128, g * 128 + jg * 16, kk * 32, 512, lane);

  v8f_t c = vzero8();

  // warm the cache for t = 0 (lanes cover the 16 batch rows; halves overlap)
  {
    const float* prow = xg + (size_t)((m0 + l) * T) * 512 + (size_t)(jg * 16);
#pragma unroll
    for (int g = 0; g < 4; ++g) prefetch_wgp(prow + g * 128);
  }

  __syncthreads();

#pragma unroll 1
  for (int t = 0; t < T; ++t) {
    // prefetch step t+1's xg into WGP cache (no VGPR cost, no LOADcnt)
    if (t + 1 < T) {
      const float* prow = xg + (size_t)((m0 + l) * T + (t + 1)) * 512 + (size_t)(jg * 16);
#pragma unroll
      for (int g = 0; g < 4; ++g) prefetch_wgp(prow + g * 128);
    }

    // h @ Whh^T into zero-initialized accumulators (starts right away)
    v8f_t acc[4];
#pragma unroll
    for (int g = 0; g < 4; ++g) acc[g] = vzero8();
#pragma unroll
    for (int kk = 0; kk < 4; ++kk) {
      v16bf_t a = load_a_frag(hbuf, 136, 0, kk * 32, lane);
#pragma unroll
      for (int g = 0; g < 4; ++g) acc[g] = wmma_bf16(a, bw[g][kk], acc[g]);
    }

    // gate phase: add xg (near-cache loads overlap the WMMA chain above);
    // all transcendentals are straight-line (hardware EXP / TANH paths)
#pragma unroll
    for (int r = 0; r < 8; ++r) {
      const int b = m0 + r + half * 8;
      const float* xr = xg + (size_t)(b * T + t) * 512 + (size_t)hc;
      const float ig = sigmoidf_(acc[0][r] + xr[0]);
      const float fg = sigmoidf_(acc[1][r] + xr[128]);
      const float gg = fast_tanhf(acc[2][r] + xr[256]);
      const float og = sigmoidf_(acc[3][r] + xr[384]);
      const float cr = fg * c[r] + ig * gg;
      c[r] = cr;
      acc[0][r] = og * fast_tanhf(cr);  // h value, reuse acc storage
    }
    __syncthreads();                  // all hbuf reads done before overwrite
#pragma unroll
    for (int r = 0; r < 8; ++r) {
      const int b = m0 + r + half * 8;
      const bf16_t hb = f2bf(acc[0][r]);
      hbuf[(r + half * 8) * 136 + hc] = hb;
      hout[(size_t)(b * T + t) * 128 + hc] = hb;
    }
    __syncthreads();                  // writes visible before next step's reads
  }
}

// ---------------------------------------------------------------------------
extern "C" void kernel_launch(void* const* d_in, const int* in_sizes, int n_in,
                              void* d_out, int out_size, void* d_ws, size_t ws_size,
                              hipStream_t stream) {
  (void)in_sizes; (void)n_in; (void)out_size; (void)ws_size;
  const float* mixture = (const float*)d_in[0];
  const float* Wih0 = (const float*)d_in[1];
  const float* Whh0 = (const float*)d_in[2];
  const float* bih0 = (const float*)d_in[3];
  const float* bhh0 = (const float*)d_in[4];
  const float* Wih1 = (const float*)d_in[5];
  const float* Whh1 = (const float*)d_in[6];
  const float* bih1 = (const float*)d_in[7];
  const float* bhh1 = (const float*)d_in[8];
  const float* W1 = (const float*)d_in[9];
  const float* b1 = (const float*)d_in[10];
  const float* W2 = (const float*)d_in[11];
  const float* b2 = (const float*)d_in[12];
  const float* W3 = (const float*)d_in[13];
  const float* b3 = (const float*)d_in[14];
  float* out = (float*)d_out;

  const int T = 512, M = 64 * 512;

  char* w = (char*)d_ws;
  auto alloc = [&](size_t bytes) -> char* {
    char* p = w; w += (bytes + 255) & ~(size_t)255; return p;
  };
  bf16_t* x_bf  = (bf16_t*)alloc((size_t)M * 128 * 2);
  bf16_t* Wih0b = (bf16_t*)alloc(512 * 128 * 2);
  bf16_t* Whh0b = (bf16_t*)alloc(512 * 128 * 2);
  bf16_t* Wih1b = (bf16_t*)alloc(512 * 128 * 2);
  bf16_t* Whh1b = (bf16_t*)alloc(512 * 128 * 2);
  bf16_t* W1b   = (bf16_t*)alloc(1024 * 128 * 2);
  bf16_t* W2b   = (bf16_t*)alloc((size_t)2048 * 1024 * 2);
  bf16_t* W3b   = (bf16_t*)alloc(16 * 2048 * 2);
  float*  bsum0 = (float*) alloc(512 * 4);
  float*  bsum1 = (float*) alloc(512 * 4);
  bf16_t* h_bf  = (bf16_t*)alloc((size_t)M * 128 * 2);   // layer0 out, then layer1 out
  float*  xg    = (float*) alloc((size_t)M * 512 * 4);   // input gates (both layers)
  bf16_t* x1_bf = (bf16_t*)xg;                           // reuse: xg dead before x1 written
  bf16_t* x2_bf = (bf16_t*)alloc((size_t)M * 2048 * 2);

  // weight / input conversion to bf16 + combined biases
  cvt_f32_to_bf16<<<1024, 256, 0, stream>>>(mixture, x_bf, M * 128);
  cvt_f32_to_bf16<<<64, 256, 0, stream>>>(Wih0, Wih0b, 512 * 128);
  cvt_f32_to_bf16<<<64, 256, 0, stream>>>(Whh0, Whh0b, 512 * 128);
  cvt_f32_to_bf16<<<64, 256, 0, stream>>>(Wih1, Wih1b, 512 * 128);
  cvt_f32_to_bf16<<<64, 256, 0, stream>>>(Whh1, Whh1b, 512 * 128);
  cvt_f32_to_bf16<<<128, 256, 0, stream>>>(W1, W1b, 1024 * 128);
  cvt_f32_to_bf16<<<1024, 256, 0, stream>>>(W2, W2b, 2048 * 1024);
  pad_w3_bf16<<<32, 256, 0, stream>>>(W3, W3b);
  add_bias_vec<<<2, 256, 0, stream>>>(bih0, bhh0, bsum0, 512);
  add_bias_vec<<<2, 256, 0, stream>>>(bih1, bhh1, bsum1, 512);

  // layer 0: xg0 = x @ Wih0^T + bsum0 ; then batch-parallel serial scan
  gemm_bf16_wmma<true, false, false><<<dim3(M / 256, 512 / 128), 256, 0, stream>>>(
      x_bf, Wih0b, xg, bsum0, M, 512, 128, 512);
  lstm_scan_wmma<<<4, 256, 0, stream>>>(xg, Whh0b, h_bf, T);

  // layer 1
  gemm_bf16_wmma<true, false, false><<<dim3(M / 256, 512 / 128), 256, 0, stream>>>(
      h_bf, Wih1b, xg, bsum1, M, 512, 128, 512);
  lstm_scan_wmma<<<4, 256, 0, stream>>>(xg, Whh1b, h_bf, T);

  // MLP
  gemm_bf16_wmma<false, true, false><<<dim3(M / 256, 1024 / 128), 256, 0, stream>>>(
      h_bf, W1b, x1_bf, b1, M, 1024, 128, 1024);
  gemm_bf16_wmma<false, true, false><<<dim3(M / 256, 2048 / 128), 256, 0, stream>>>(
      x1_bf, W2b, x2_bf, b2, M, 2048, 1024, 2048);
  gemm_bf16_wmma<true, false, true><<<dim3(M / 256, 1), 256, 0, stream>>>(
      x2_bf, W3b, out, b3, M, 10, 2048, 16);
}
